// TemporalInteractionModule_17051020165231
// MI455X (gfx1250) — 8-bit matrix operand order repaired and checked by reading, compiled, not run on hardware
//
#include <hip/hip_runtime.h>

#define T_STEPS 512
#define BATCH   64
#define CHAN    512
#define VTH     1.0f
#define ALPHA   0.6f

typedef __attribute__((ext_vector_type(16))) int   v16i;
typedef __attribute__((ext_vector_type(8)))  float v8f;

union V16 {
  v16i v;
  uint4 q[4];
  unsigned long long d[8];
};

__device__ __forceinline__ unsigned f32_to_e4m3_bl(float f) {
  unsigned u   = __float_as_uint(f);
  unsigned sgn = (u >> 24) & 0x80u;
  unsigned a   = u & 0x7FFFFFFFu;
  unsigned r   = a + 0x0007FFFFu + ((a >> 20) & 1u);
  unsigned em  = (r >> 20) - 960u;
  unsigned den = (unsigned)(__uint_as_float(a) * 512.0f + 0.5f);
  unsigned res = (a < 0x3C800000u) ? den : em;
  res = (res > 0x7Eu) ? 0x7Eu : res;
  return sgn | res;
}

__device__ __forceinline__ int pack2_e4m3(float v0, float v1) {
#if __has_builtin(__builtin_amdgcn_cvt_pk_fp8_f32)
  return __builtin_amdgcn_cvt_pk_fp8_f32(v0, v1, 0, false);
#else
  return (int)(f32_to_e4m3_bl(v0) | (f32_to_e4m3_bl(v1) << 8));
#endif
}

__global__ void snn_prep(const float* __restrict__ conv_w, const float* __restrict__ conv_b,
                         const float* __restrict__ gamma,  const float* __restrict__ beta,
                         const float* __restrict__ mean,   const float* __restrict__ var,
                         unsigned char* __restrict__ w8, float* __restrict__ scale,
                         float* __restrict__ bias) {
  int i = blockIdx.x * 256 + threadIdx.x;
  if (i < CHAN * CHAN) {
    int d = i >> 9;
    int c = i & (CHAN - 1);
    w8[i] = (unsigned char)f32_to_e4m3_bl(conv_w[(d * CHAN + c) * 3 + 1]);
  }
  if (i < CHAN) {
    float is = gamma[i] * rsqrtf(var[i] + 1e-5f);
    scale[i] = is * 0.1f;
    bias[i]  = ((conv_b[i] - mean[i]) * is + beta[i]) * 0.1f;
  }
}

__global__ __launch_bounds__(512, 1)
void snn_persistent(const float* __restrict__ x, const unsigned char* __restrict__ w8,
                    const float* __restrict__ scale, const float* __restrict__ bias,
                    float* __restrict__ out) {
  __shared__ unsigned char prevbuf[2][16 * CHAN];

  const int b0   = blockIdx.x * 16;
  const int wave = threadIdx.x >> 5;
  const int lane = threadIdx.x & 31;
  const int lh   = lane & 15;
  const int hi   = lane >> 4;

  V16  wB[2][4];
  float sc[2], bi[2];
#pragma unroll
  for (int tt = 0; tt < 2; ++tt) {
    const int n = wave * 32 + tt * 16 + lh;
    sc[tt] = scale[n];
    bi[tt] = bias[n];
#pragma unroll
    for (int kc = 0; kc < 4; ++kc)
#pragma unroll
      for (int g = 0; g < 4; ++g)
        wB[tt][kc].q[g] = *(const uint4*)(w8 + n * CHAN + kc * 128 + g * 32 + hi * 16);
  }

  float vst[2][8];
#pragma unroll
  for (int tt = 0; tt < 2; ++tt)
#pragma unroll
    for (int r = 0; r < 8; ++r) vst[tt][r] = 0.0f;

  for (int i = threadIdx.x * 4; i < 2 * 16 * CHAN; i += 512 * 4)
    *(unsigned int*)(&prevbuf[0][0] + i) = 0u;
  __syncthreads();

  int cur = 0;
  for (int t = 0; t < T_STEPS; ++t) {
    {
      int tp = (t + 4 < T_STEPS) ? (t + 4) : (T_STEPS - 1);
      __builtin_prefetch(x + ((size_t)tp * BATCH + b0) * CHAN + threadIdx.x * 16, 0, 1);
    }

    const size_t gbase = ((size_t)t * BATCH + b0) * CHAN;
    float xr[2][8];
#pragma unroll
    for (int tt = 0; tt < 2; ++tt) {
      const int n = wave * 32 + tt * 16 + lh;
#pragma unroll
      for (int r = 0; r < 8; ++r)
        xr[tt][r] = x[gbase + (size_t)(r + hi * 8) * CHAN + n];
    }

    const unsigned char* pc = &prevbuf[cur][0];
    v8f acc[2] = {{0,0,0,0,0,0,0,0}, {0,0,0,0,0,0,0,0}};
#pragma unroll
    for (int kc = 0; kc < 4; ++kc) {
      V16 A;
#pragma unroll
      for (int h2 = 0; h2 < 2; ++h2)
#pragma unroll
        for (int qd = 0; qd < 4; ++qd)
          A.d[h2 * 4 + qd] = *(const unsigned long long*)
              (pc + lh * CHAN + kc * 128 + h2 * 64 + (qd >> 1) * 32 + hi * 16 + (qd & 1) * 8);

      acc[0] = __builtin_amdgcn_wmma_f32_16x16x128_fp8_fp8(A.v, wB[0][kc].v, (short)0, acc[0], false, false);
      acc[1] = __builtin_amdgcn_wmma_f32_16x16x128_fp8_fp8(A.v, wB[1][kc].v, (short)0, acc[1], false, false);
    }

    unsigned char* pn = &prevbuf[cur ^ 1][0];
#pragma unroll
    for (int tt = 0; tt < 2; ++tt) {
      const int n = wave * 32 + tt * 16 + lh;
#pragma unroll
      for (int r = 0; r < 8; r += 2) {
        float nv01[2];
#pragma unroll
        for (int j = 0; j < 2; ++j) {
          const int rr = r + j;
          float pv = fmaf(vst[tt][rr], 0.9f, bi[tt]);
          pv = fmaf(acc[tt][rr], sc[tt], pv);
          bool sp = pv >= VTH;
          vst[tt][rr] = sp ? 0.0f : pv;
          float nv = fmaf(xr[tt][rr], 1.0f - ALPHA, sp ? ALPHA : 0.0f);
          out[gbase + (size_t)(rr + hi * 8) * CHAN + n] = nv;
          nv01[j] = nv;
        }
        int p = pack2_e4m3(nv01[0], nv01[1]);
        pn[(r + hi * 8) * CHAN + n]     = (unsigned char)p;
        pn[(r + 1 + hi * 8) * CHAN + n] = (unsigned char)(p >> 8);
      }
    }
    __syncthreads();
    cur ^= 1;
  }
}

extern "C" void kernel_launch(void* const* d_in, const int* in_sizes, int n_in,
                              void* d_out, int out_size, void* d_ws, size_t ws_size,
                              hipStream_t stream) {
  const float* x      = (const float*)d_in[0];
  const float* conv_w = (const float*)d_in[1];
  const float* conv_b = (const float*)d_in[2];
  const float* gamma  = (const float*)d_in[3];
  const float* beta   = (const float*)d_in[4];
  const float* mean   = (const float*)d_in[5];
  const float* var    = (const float*)d_in[6];
  float* out = (float*)d_out;

  unsigned char* w8 = (unsigned char*)d_ws;
  float* scale = (float*)((char*)d_ws + CHAN * CHAN);
  float* bias  = (float*)((char*)d_ws + CHAN * CHAN + CHAN * 4);

  snn_prep<<<dim3((CHAN * CHAN + 255) / 256), dim3(256), 0, stream>>>(
      conv_w, conv_b, gamma, beta, mean, var, w8, scale, bias);

  snn_persistent<<<dim3(BATCH / 16), dim3(512), 0, stream>>>(x, w8, scale, bias, out);
}
